// EquivariantBlock_44736379355272
// MI455X (gfx1250) — matrix operand order repaired and checked by reading, not compiled, not run on hardware
//
#include <hip/hip_runtime.h>
#include <cmath>
#include <complex>
#include <cstring>

// ---------------------------------------------------------------------------
// CDNA5 (gfx1250) equivariant GNN block.
//   - radial MLP [E,32]x[32,648] -> v_wmma_f32_16x16x32_f16 (the only
//     GEMM-shaped 4.15 GFLOP of the workload)
//   - CG tensor product: unrolled VALU, CG tables in LDS
//   - scatter: global_atomic_add_f32
// ---------------------------------------------------------------------------

typedef __attribute__((ext_vector_type(16))) _Float16 v16h;
typedef __attribute__((ext_vector_type(8)))  float    v8f;

#define DIM     66
#define NTILES  41          // ceil(648/16) column tiles of rad_w2
#define WST     660         // padded LDS stride for per-edge radial weights
#define CGTOT   540         // total dense CG floats (all 14 keys)

__device__ __forceinline__ float sigmoidf_(float x) { return 1.f / (1.f + __expf(-x)); }
__device__ __forceinline__ float siluf_(float x)    { return x / (1.f + __expf(-x)); }

// A-fragment K index for 16-bit 16x32 A (matches B): lanes<16 hold K 0-15;
// lanes>=16 hold K 16-31.
__device__ __forceinline__ int kA_(int h, int hi) { return h + (hi ? 16 : 0); }
// B-fragment K index (contiguous order, same as kA_): lanes<16 hold K=0..15 sequentially,
// lanes>=16 hold K=16..31.
__device__ __forceinline__ int kB_(int h, int hi) { return h + (hi ? 16 : 0); }

// ---------------------------------------------------------------------------
// Pre-convert rad_w2 (f32 [32][648]) into f16 B-fragments, scaled by 1/sqrt(32),
// zero padded to 41 tiles of 16 columns.  frag[tile*512 + lane*16 + h].
// ---------------------------------------------------------------------------
__global__ void prep_w2_kernel(const float* __restrict__ rad_w2, _Float16* __restrict__ frag)
{
    const int t    = blockIdx.x;
    const int lane = threadIdx.x;       // 0..31
    const int hi   = (lane >= 16) ? 1 : 0;
    const int col  = lane & 15;
    const float s  = 0.17677669529663687f;   // 1/sqrt(32)
#pragma unroll
    for (int h = 0; h < 16; ++h) {
        int   k = kB_(h, hi);
        int   n = t * 16 + col;
        float v = (n < 648) ? rad_w2[k * 648 + n] * s : 0.f;
        frag[(size_t)t * 512 + lane * 16 + h] = (_Float16)v;
    }
}

// ---------------------------------------------------------------------------
// Block-diagonal irreps linear on one 66-dim row.
// groups: (12,l0)(4,l1)(4,l1)(6,l2) at offsets 0,12,24,36
// ---------------------------------------------------------------------------
__device__ __forceinline__ void irreps_row_(const float* __restrict__ x,
                                            const float* __restrict__ w0,
                                            const float* __restrict__ w1o,
                                            const float* __restrict__ w1e,
                                            const float* __restrict__ w2,
                                            const float* __restrict__ b0,
                                            float* __restrict__ o)
{
#pragma unroll
    for (int v = 0; v < 12; ++v) {                      // l=0, 1/sqrt(12), +bias
        float s = 0.f;
#pragma unroll
        for (int u = 0; u < 12; ++u) s += x[u] * w0[u * 12 + v];
        o[v] = s * 0.28867513459481287f + b0[v];
    }
#pragma unroll
    for (int v = 0; v < 4; ++v)                         // l=1 odd, 1/sqrt(4)
#pragma unroll
        for (int i = 0; i < 3; ++i) {
            float s = 0.f;
#pragma unroll
            for (int u = 0; u < 4; ++u) s += x[12 + u * 3 + i] * w1o[u * 4 + v];
            o[12 + v * 3 + i] = s * 0.5f;
        }
#pragma unroll
    for (int v = 0; v < 4; ++v)                         // l=1 even
#pragma unroll
        for (int i = 0; i < 3; ++i) {
            float s = 0.f;
#pragma unroll
            for (int u = 0; u < 4; ++u) s += x[24 + u * 3 + i] * w1e[u * 4 + v];
            o[24 + v * 3 + i] = s * 0.5f;
        }
#pragma unroll
    for (int v = 0; v < 6; ++v)                         // l=2, 1/sqrt(6)
#pragma unroll
        for (int i = 0; i < 5; ++i) {
            float s = 0.f;
#pragma unroll
            for (int u = 0; u < 6; ++u) s += x[36 + u * 5 + i] * w2[u * 6 + v];
            o[36 + v * 5 + i] = s * 0.4082482904638631f;
        }
}

__global__ void node_kernel(const float* __restrict__ h,
                            const float* __restrict__ lw0, const float* __restrict__ lw1o,
                            const float* __restrict__ lw1e, const float* __restrict__ lw2,
                            const float* __restrict__ lb0,
                            const float* __restrict__ rw0, const float* __restrict__ rw1o,
                            const float* __restrict__ rw1e, const float* __restrict__ rw2,
                            const float* __restrict__ rb0,
                            float* __restrict__ h_in, float* __restrict__ agg, int NN)
{
    int n = blockIdx.x * blockDim.x + threadIdx.x;
    if (n >= NN) return;
    float x[DIM], o[DIM];
    const float* hr = h + (size_t)n * DIM;
#pragma unroll
    for (int i = 0; i < DIM; ++i) x[i] = hr[i];

    irreps_row_(x, lw0, lw1o, lw1e, lw2, lb0, o);       // h_in for messages
    float* hp = h_in + (size_t)n * DIM;
#pragma unroll
    for (int i = 0; i < DIM; ++i) hp[i] = o[i];

    irreps_row_(x, rw0, rw1o, rw1e, rw2, rb0, o);       // residual -> agg init
    float* ap = agg + (size_t)n * DIM;
#pragma unroll
    for (int i = 0; i < DIM; ++i) ap[i] = o[i];
}

// ---------------------------------------------------------------------------
// One CG path:  out[v][k] += sum_u w[u][v] * sum_i xs[u][i] * G[i][k],
// G[i][k] = sum_j Y[j] * CG[i][j][k].  Fully unrolled; cg from LDS.
// ---------------------------------------------------------------------------
template <int MUL1, int D1, int D2, int MULO, int DO>
__device__ __forceinline__ void tp_path(const float* __restrict__ xs,
                                        const float* __restrict__ Yv,
                                        const float* __restrict__ cg,
                                        const float* __restrict__ w,
                                        float* __restrict__ out)
{
    float G[D1 * DO];
#pragma unroll
    for (int i = 0; i < D1 * DO; ++i) G[i] = 0.f;
#pragma unroll
    for (int j = 0; j < D2; ++j) {
        float yv = Yv[j];
#pragma unroll
        for (int i = 0; i < D1; ++i)
#pragma unroll
            for (int k = 0; k < DO; ++k)
                G[i * DO + k] += cg[(i * D2 + j) * DO + k] * yv;
    }
#pragma unroll
    for (int u = 0; u < MUL1; ++u) {
        float t[DO];
#pragma unroll
        for (int k = 0; k < DO; ++k) {
            float s = 0.f;
#pragma unroll
            for (int i = 0; i < D1; ++i) s += xs[u * D1 + i] * G[i * DO + k];
            t[k] = s;
        }
#pragma unroll
        for (int v = 0; v < MULO; ++v) {
            float wv = w[u * MULO + v];
#pragma unroll
            for (int k = 0; k < DO; ++k) out[v * DO + k] += wv * t[k];
        }
    }
}

// ---------------------------------------------------------------------------
// Edge kernel: 1 wave (32 threads) per workgroup, 16 edges per workgroup.
// Phase A: geometry+Y; Phase B: radial MLP via WMMA into LDS; Phase C:
// tensor product (2 threads/edge, split by output irrep) + atomic scatter.
// CG dense offsets (key -> off): 000:0 011:1 022:10 101:35 110:44 111:53
// 112:80 121:125 122:170 202:245 211:270 220:315 221:340 222:415
// ---------------------------------------------------------------------------
__global__ __launch_bounds__(32) void edge_kernel(
    const int* __restrict__ e_src, const int* __restrict__ e_dst,
    const float* __restrict__ e_attr, const float* __restrict__ rad_w1,
    const float* __restrict__ h_in, const _Float16* __restrict__ w2frag,
    const float* __restrict__ cg_g, float* __restrict__ agg,
    int N, int E, int Etot)
{
    extern __shared__ float smem[];
    float* w_s  = smem;                 // [16][WST]  per-edge radial weights
    float* Y_s  = w_s + 16 * WST;       // [16][12]   sph harmonics (9 used)
    float* rn_s = Y_s + 16 * 12;        // [16]
    float* cg_s = rn_s + 16;            // [CGTOT]

    const int tid  = threadIdx.x;       // 0..31, single wave
    const int hi   = (tid >= 16) ? 1 : 0;
    const int lo16 = tid & 15;
    const int e0   = blockIdx.x * 16;

    for (int i = tid; i < CGTOT; i += 32) cg_s[i] = cg_g[i];

    if (tid < 16) {
        int   eg = e0 + tid;
        float ax = 0.f, ay = 0.f, az = 0.f;
        if (eg < Etot) {
            ax = e_attr[eg * 3 + 0];
            ay = e_attr[eg * 3 + 1];
            az = e_attr[eg * 3 + 2];
            int b    = eg / E;
            int node = b * N + e_src[eg];
            __builtin_prefetch(h_in + (size_t)node * DIM, 0, 0);   // global_prefetch_b8
        }
        float rn = sqrtf(ax * ax + ay * ay + az * az) + 1e-8f;
        float x = ax / rn, y = ay / rn, z = az / rn;
        if (eg >= Etot) { rn = 0.f; x = 0.f; y = 0.f; z = 0.f; }   // hidden -> 0
        rn_s[tid] = rn;
        const float s3 = 1.7320508075688772f, s5 = 2.23606797749979f, s15 = 3.872983346207417f;
        float* Yp = Y_s + tid * 12;
        Yp[0] = 1.f;
        Yp[1] = s3 * y;  Yp[2] = s3 * z;  Yp[3] = s3 * x;
        Yp[4] = s15 * x * y;  Yp[5] = s15 * y * z;
        Yp[6] = 0.5f * s5 * (3.f * z * z - 1.f);
        Yp[7] = s15 * x * z;  Yp[8] = 0.5f * s15 * (x * x - y * y);
    }
    __syncthreads();

    // -------- Phase B: w[16 edges][648] = silu(rn*w1) @ w2  via WMMA ------
    {
        const float rn = rn_s[lo16];
        v16h a;
#pragma unroll
        for (int h = 0; h < 16; ++h) {
            int   k  = kA_(h, hi);
            float xx = rn * rad_w1[k];
            a[h] = (_Float16)siluf_(xx);
        }
#pragma unroll 4
        for (int t = 0; t < NTILES; ++t) {
            v16h b = *reinterpret_cast<const v16h*>(w2frag + (size_t)t * 512 + tid * 16);
            v8f  c = {};
            c = __builtin_amdgcn_wmma_f32_16x16x32_f16(false, a, false, b, (short)0, c,
                                                       false, false);
#pragma unroll
            for (int r = 0; r < 8; ++r) {
                int m = hi ? (r + 8) : r;            // ISA C-layout
                w_s[(m)*WST + t * 16 + lo16] = c[r];
            }
        }
    }
    __syncthreads();

    // -------- Phase C: tensor product + scatter (2 threads per edge) ------
    const int le   = tid >> 1;
    const int half = tid & 1;
    const int eg   = e0 + le;
    if (eg >= Etot) return;

    const int    b      = eg / E;
    const int    node_s = b * N + e_src[eg];
    const int    node_d = b * N + e_dst[eg];
    const float* xr     = h_in + (size_t)node_s * DIM;
    const float* Yp     = Y_s + le * 12;
    const float* wr     = w_s + le * WST;
    float*       ag     = agg + (size_t)node_d * DIM;

    if (half == 0) {
        // ---- output group c=3 (6 x l=2), fan-weighted cost ~= other half ----
        float out3[30];
#pragma unroll
        for (int i = 0; i < 30; ++i) out3[i] = 0.f;
        tp_path<12, 1, 5, 6, 5>(xr + 0,  Yp + 4, cg_s + 10,  wr + 192, out3);  // (0,2,3)
        tp_path<4,  3, 3, 6, 5>(xr + 12, Yp + 1, cg_s + 80,  wr + 344, out3);  // (1,1,3)
        tp_path<4,  3, 5, 6, 5>(xr + 24, Yp + 4, cg_s + 170, wr + 432, out3);  // (2,2,3)
        tp_path<6,  5, 1, 6, 5>(xr + 36, Yp + 0, cg_s + 245, wr + 456, out3);  // (3,0,3)
        tp_path<6,  5, 5, 6, 5>(xr + 36, Yp + 4, cg_s + 415, wr + 612, out3);  // (3,2,3)
        const float a3 = 0.39528470752104744f;   // sqrt(5/32)
#pragma unroll
        for (int i = 0; i < 30; ++i) unsafeAtomicAdd(ag + 36 + i, a3 * out3[i]);
    } else {
        // ---- output groups c=0,1,2 ----
        float out0[12], out1[12], out2[12];
#pragma unroll
        for (int i = 0; i < 12; ++i) { out0[i] = 0.f; out1[i] = 0.f; out2[i] = 0.f; }
        tp_path<12, 1, 1, 12, 1>(xr + 0,  Yp + 0, cg_s + 0,   wr + 0,   out0); // (0,0,0)
        tp_path<4,  3, 3, 12, 1>(xr + 12, Yp + 1, cg_s + 44,  wr + 280, out0); // (1,1,0)
        tp_path<6,  5, 5, 12, 1>(xr + 36, Yp + 4, cg_s + 315, wr + 516, out0); // (3,2,0)
        tp_path<12, 1, 3, 4, 3>(xr + 0,  Yp + 1, cg_s + 1,   wr + 144, out1);  // (0,1,1)
        tp_path<4,  3, 1, 4, 3>(xr + 12, Yp + 0, cg_s + 35,  wr + 264, out1);  // (1,0,1)
        tp_path<4,  3, 5, 4, 3>(xr + 12, Yp + 4, cg_s + 125, wr + 368, out1);  // (1,2,1)
        tp_path<4,  3, 3, 4, 3>(xr + 24, Yp + 1, cg_s + 53,  wr + 400, out1);  // (2,1,1)
        tp_path<6,  5, 3, 4, 3>(xr + 36, Yp + 1, cg_s + 270, wr + 492, out1);  // (3,1,1)
        tp_path<4,  3, 3, 4, 3>(xr + 12, Yp + 1, cg_s + 53,  wr + 328, out2);  // (1,1,2)
        tp_path<4,  3, 1, 4, 3>(xr + 24, Yp + 0, cg_s + 35,  wr + 384, out2);  // (2,0,2)
        tp_path<4,  3, 5, 4, 3>(xr + 24, Yp + 4, cg_s + 125, wr + 416, out2);  // (2,2,2)
        tp_path<6,  5, 5, 4, 3>(xr + 36, Yp + 4, cg_s + 340, wr + 588, out2);  // (3,2,2)
        const float a0 = 0.2132007163556104f;    // 1/sqrt(22)
        const float a1 = 0.31622776601683794f;   // sqrt(3/30)
        const float a2 = 0.4082482904638631f;    // sqrt(3/18)
#pragma unroll
        for (int i = 0; i < 12; ++i) unsafeAtomicAdd(ag + i,      a0 * out0[i]);
#pragma unroll
        for (int i = 0; i < 12; ++i) unsafeAtomicAdd(ag + 12 + i, a1 * out1[i]);
#pragma unroll
        for (int i = 0; i < 12; ++i) unsafeAtomicAdd(ag + 24 + i, a2 * out2[i]);
    }
}

// ---------------------------------------------------------------------------
// norm_act:  out = x * sigmoid(||x_irrep||), norm per irrep copy   (silu(n)/n == sigmoid(n))
// ---------------------------------------------------------------------------
__global__ void final_kernel(const float* __restrict__ agg, float* __restrict__ out, int NN)
{
    int n = blockIdx.x * blockDim.x + threadIdx.x;
    if (n >= NN) return;
    const float* a = agg + (size_t)n * DIM;
    float*       o = out + (size_t)n * DIM;
#pragma unroll
    for (int u = 0; u < 12; ++u) {
        float v = a[u];
        o[u] = v * sigmoidf_(sqrtf(v * v + 1e-12f));
    }
#pragma unroll
    for (int u = 0; u < 8; ++u) {                       // both l=1 groups
        int   base = 12 + u * 3;
        float v0 = a[base], v1 = a[base + 1], v2 = a[base + 2];
        float s  = sigmoidf_(sqrtf(v0 * v0 + v1 * v1 + v2 * v2 + 1e-12f));
        o[base] = v0 * s; o[base + 1] = v1 * s; o[base + 2] = v2 * s;
    }
#pragma unroll
    for (int u = 0; u < 6; ++u) {                       // l=2 group
        int   base = 36 + u * 5;
        float s2 = 1e-12f;
#pragma unroll
        for (int k = 0; k < 5; ++k) { float v = a[base + k]; s2 += v * v; }
        float s = sigmoidf_(sqrtf(s2));
#pragma unroll
        for (int k = 0; k < 5; ++k) o[base + k] = a[base + k] * s;
    }
}

// ---------------------------------------------------------------------------
// Host-side real Clebsch-Gordan construction (exact port of the reference).
// ---------------------------------------------------------------------------
namespace cgbuild {

static double factd(int n) { double r = 1.0; for (int i = 2; i <= n; ++i) r *= i; return r; }

static double su2cg(int j1, int j2, int j3, int m1, int m2, int m3)
{
    if (m1 + m2 != m3) return 0.0;
    double pre = std::sqrt((2.0 * j3 + 1.0) * factd(j3 + j1 - j2) * factd(j3 - j1 + j2) *
                           factd(j1 + j2 - j3) / factd(j1 + j2 + j3 + 1));
    pre *= std::sqrt(factd(j3 + m3) * factd(j3 - m3) * factd(j1 - m1) * factd(j1 + m1) *
                     factd(j2 - m2) * factd(j2 + m2));
    double s = 0.0;
    for (int k = 0; k <= j1 + j2 - j3; ++k) {
        int a0 = k, a1 = j1 + j2 - j3 - k, a2 = j1 - m1 - k, a3 = j2 + m2 - k;
        int a4 = j3 - j2 + m1 + k, a5 = j3 - j1 - m2 + k;
        if (a0 < 0 || a1 < 0 || a2 < 0 || a3 < 0 || a4 < 0 || a5 < 0) continue;
        double d = factd(a0) * factd(a1) * factd(a2) * factd(a3) * factd(a4) * factd(a5);
        s += ((k & 1) ? -1.0 : 1.0) / d;
    }
    return pre * s;
}

static void u_real(int l, std::complex<double> U[5][5])
{
    for (int i = 0; i < 5; ++i)
        for (int j = 0; j < 5; ++j) U[i][j] = 0.0;
    const double sq = 1.0 / std::sqrt(2.0);
    for (int m = -l; m <= l; ++m) {
        int i = m + l;
        if (m > 0) {
            U[i][l + m] = ((m & 1) ? -1.0 : 1.0) * sq;
            U[i][l - m] = sq;
        } else if (m == 0) {
            U[i][l] = 1.0;
        } else {
            int    mm  = -m;
            double sgn = (mm & 1) ? -1.0 : 1.0;       // (-1)^m
            U[i][l + m] = std::complex<double>(0.0, sq);
            U[i][l - m] = std::complex<double>(0.0, -sgn * sq);
        }
    }
}

static void real_cg(int l1, int l2, int l3, float* out)
{
    const int d1 = 2 * l1 + 1, d2 = 2 * l2 + 1, d3 = 2 * l3 + 1;
    std::complex<double> Cc[5][5][5];
    for (int i = 0; i < 5; ++i)
        for (int j = 0; j < 5; ++j)
            for (int k = 0; k < 5; ++k) Cc[i][j][k] = 0.0;
    for (int m1 = -l1; m1 <= l1; ++m1)
        for (int m2 = -l2; m2 <= l2; ++m2) {
            int m3 = m1 + m2;
            if (m3 >= -l3 && m3 <= l3)
                Cc[l1 + m1][l2 + m2][l3 + m3] = su2cg(l1, l2, l3, m1, m2, m3);
        }
    std::complex<double> U1[5][5], U2[5][5], U3[5][5];
    u_real(l1, U1); u_real(l2, U2); u_real(l3, U3);
    double Tre[125], Tim[125], sre = 0.0, sim = 0.0;
    for (int a = 0; a < d1; ++a)
        for (int bq = 0; bq < d2; ++bq)
            for (int c = 0; c < d3; ++c) {
                std::complex<double> t(0.0, 0.0);
                for (int i = 0; i < d1; ++i)
                    for (int j = 0; j < d2; ++j)
                        for (int k = 0; k < d3; ++k)
                            t += std::conj(U1[a][i]) * std::conj(U2[bq][j]) * U3[c][k] * Cc[i][j][k];
                int idx = (a * d2 + bq) * d3 + c;
                Tre[idx] = t.real(); Tim[idx] = t.imag();
                sre += std::fabs(t.real()); sim += std::fabs(t.imag());
            }
    const double* T = (sre >= sim) ? Tre : Tim;
    double nrm = 0.0;
    for (int i = 0; i < d1 * d2 * d3; ++i) nrm += T[i] * T[i];
    nrm = std::sqrt(nrm);
    for (int i = 0; i < d1 * d2 * d3; ++i) out[i] = (float)(T[i] / nrm);
}

} // namespace cgbuild

// ---------------------------------------------------------------------------
extern "C" void kernel_launch(void* const* d_in, const int* in_sizes, int n_in,
                              void* d_out, int out_size, void* d_ws, size_t ws_size,
                              hipStream_t stream)
{
    const float* h      = (const float*)d_in[0];
    const int*   e_src  = (const int*)d_in[1];
    const int*   e_dst  = (const int*)d_in[2];
    const float* e_attr = (const float*)d_in[3];
    const float* lw0    = (const float*)d_in[4];
    const float* lw1o   = (const float*)d_in[5];
    const float* lw1e   = (const float*)d_in[6];
    const float* lw2    = (const float*)d_in[7];
    const float* lb0    = (const float*)d_in[8];
    const float* rw0    = (const float*)d_in[9];
    const float* rw1o   = (const float*)d_in[10];
    const float* rw1e   = (const float*)d_in[11];
    const float* rw2    = (const float*)d_in[12];
    const float* rb0    = (const float*)d_in[13];
    const float* rad_w1 = (const float*)d_in[14];
    const float* rad_w2 = (const float*)d_in[15];

    const int B    = 2;
    const int Etot = in_sizes[1];          // B*E
    const int E    = Etot / B;
    const int NN   = in_sizes[0] / DIM;    // B*N
    const int N    = NN / B;

    // workspace layout (all fully written before use)
    float*    h_in = (float*)d_ws;
    float*    agg  = h_in + (size_t)NN * DIM;
    float*    cg_g = agg + (size_t)NN * DIM;
    _Float16* frag = (_Float16*)(cg_g + 576);  // keeps 32B alignment for v16h loads

    // dense real-CG tables, key offsets must match device literals:
    // 000:0 011:1 022:10 101:35 110:44 111:53 112:80 121:125 122:170
    // 202:245 211:270 220:315 221:340 222:415   (total 540)
    struct KD { int l1, l2, l3, off; };
    static const KD kd[14] = {
        {0,0,0,0},  {0,1,1,1},   {0,2,2,10},  {1,0,1,35},  {1,1,0,44},
        {1,1,1,53}, {1,1,2,80},  {1,2,1,125}, {1,2,2,170}, {2,0,2,245},
        {2,1,1,270},{2,2,0,315}, {2,2,1,340}, {2,2,2,415}};
    static float hcg[CGTOT];               // deterministic, refilled identically each call
    for (int i = 0; i < 14; ++i) cgbuild::real_cg(kd[i].l1, kd[i].l2, kd[i].l3, hcg + kd[i].off);
    hipMemcpyAsync(cg_g, hcg, CGTOT * sizeof(float), hipMemcpyHostToDevice, stream);

    prep_w2_kernel<<<NTILES, 32, 0, stream>>>(rad_w2, frag);

    node_kernel<<<(NN + 255) / 256, 256, 0, stream>>>(h, lw0, lw1o, lw1e, lw2, lb0,
                                                      rw0, rw1o, rw1e, rw2, rb0,
                                                      h_in, agg, NN);

    const size_t smem_bytes = (size_t)(16 * WST + 16 * 12 + 16 + CGTOT) * sizeof(float);
    edge_kernel<<<(Etot + 15) / 16, 32, smem_bytes, stream>>>(e_src, e_dst, e_attr, rad_w1,
                                                              h_in, frag, cg_g, agg,
                                                              N, E, Etot);

    final_kernel<<<(NN + 255) / 256, 256, 0, stream>>>(agg, (float*)d_out, NN);
}
